// SSMBlock_56521769615371
// MI455X (gfx1250) — matrix operand order repaired and checked by reading, not compiled, not run on hardware
//
#include <hip/hip_runtime.h>
#include <hip/hip_bf16.h>

// ---------------- problem constants (from reference) ----------------
#define BD      4
#define SD      2048
#define DMODEL  1024
#define DINNER  2048
#define NROWS   (BD * SD)          // 8192 token rows
#define DSTATE  16
#define DCONV   4
#define DECAYF  0.95f
#define LNEPS   1e-5f

#define SCHUNK  256                // scan chunk length
#define NCHUNK  (SD / SCHUNK)      // 8 chunks per sequence
#define LOG2DECAY (-0.07400058144377693f)   // log2(0.95)

typedef __bf16 bf16;
typedef __attribute__((ext_vector_type(16))) __bf16 v16bf;
typedef __attribute__((ext_vector_type(8)))  float  v8f;

// POD 16-byte chunk (HIP's int4 has ctors -> not union-safe)
struct alignas(16) B128 { unsigned int d[4]; };
union Frag { v16bf v; B128 q[2]; };

__device__ __forceinline__ float silu_f(float v) {
    return v / (1.0f + __expf(-v));
}

// ---------------- 0) f32 -> bf16 convert ----------------
__global__ void cvt_f32_bf16_kernel(const float* __restrict__ src,
                                    bf16* __restrict__ dst, size_t n) {
    size_t i = (size_t)blockIdx.x * blockDim.x + threadIdx.x;
    if (i < n) dst[i] = (bf16)src[i];
}

// ---------------- 0b) Bexp = exp(B_log) ----------------
__global__ void exp_kernel(const float* __restrict__ src,
                           float* __restrict__ dst, int n) {
    int i = blockIdx.x * blockDim.x + threadIdx.x;
    if (i < n) dst[i] = __expf(src[i]);
}

// ---------------- 1) WMMA bf16 GEMM: C[M,N] = A[M,K] * W[N,K]^T ----------------
// Block: 256 threads = 8 waves. Block tile 128(M) x 128(N), K step 32.
// Waves arranged 4(M) x 2(N); each wave owns a 32x64 strip -> 8 WMMAs / K-step.
// Double-buffered LDS filled by GLOBAL_LOAD_ASYNC_TO_LDS_B128 DMA (ASYNCcnt):
// the async fill of tile k+1 runs while the 8 WMMAs of tile k execute; each wave
// waits its own ASYNCcnt, then the barrier publishes the buffer. No data VGPRs.
__launch_bounds__(256)
__global__ void gemm_bf16_wmma_kernel(const bf16* __restrict__ A,
                                      const bf16* __restrict__ W,
                                      float* __restrict__ C,
                                      int Msz, int Nsz, int Ksz) {
    __shared__ bf16 As[2][128][40];   // +8 pad: 80B row stride (16B aligned)
    __shared__ bf16 Bs[2][128][40];
    constexpr unsigned BUFB = 128u * 40u * (unsigned)sizeof(bf16);  // 10240 B

    const int tid  = threadIdx.x;
    const int lane = tid & 31;
    const int wave = tid >> 5;
    const int wm   = wave & 3;        // 4 M strips of 32
    const int wn   = wave >> 2;       // 2 N strips of 64
    const int half = lane >> 4;       // 0: lanes 0-15, 1: lanes 16-31
    const int l16  = lane & 15;
    const int blockM = blockIdx.y * 128;
    const int blockN = blockIdx.x * 128;

    // staging geometry: 1024 16B chunks per (A+B) 128x32 tile pair -> 4 per thread
    const int srow0 = tid >> 2;                 // rows 0..63
    const int srow1 = srow0 + 64;               // rows 64..127
    const int skc   = (tid & 3) * 8;

    // running global source pointers (bumped 32 elements = 64B per K-step)
    const bf16* gA0 = A + (size_t)(blockM + srow0) * Ksz + skc;
    const bf16* gA1 = A + (size_t)(blockM + srow1) * Ksz + skc;
    const bf16* gB0 = W + (size_t)(blockN + srow0) * Ksz + skc;
    const bf16* gB1 = W + (size_t)(blockN + srow1) * Ksz + skc;

    // LDS destination byte offsets (buffer 0); buffer 1 = +BUFB
    const unsigned lA0 = (unsigned)(size_t)&As[0][srow0][skc];
    const unsigned lA1 = (unsigned)(size_t)&As[0][srow1][skc];
    const unsigned lB0 = (unsigned)(size_t)&Bs[0][srow0][skc];
    const unsigned lB1 = (unsigned)(size_t)&Bs[0][srow1][skc];

    auto stage_async = [&](unsigned bufOff) {
        asm volatile("global_load_async_to_lds_b128 %0, %1, off"
                     :: "v"(lA0 + bufOff), "v"(gA0) : "memory");
        asm volatile("global_load_async_to_lds_b128 %0, %1, off"
                     :: "v"(lA1 + bufOff), "v"(gA1) : "memory");
        asm volatile("global_load_async_to_lds_b128 %0, %1, off"
                     :: "v"(lB0 + bufOff), "v"(gB0) : "memory");
        asm volatile("global_load_async_to_lds_b128 %0, %1, off"
                     :: "v"(lB1 + bufOff), "v"(gB1) : "memory");
        gA0 += 32; gA1 += 32; gB0 += 32; gB1 += 32;
    };

    v8f acc[2][4];
#pragma unroll
    for (int i = 0; i < 2; ++i)
#pragma unroll
        for (int j = 0; j < 4; ++j)
#pragma unroll
            for (int r = 0; r < 8; ++r) acc[i][j][r] = 0.0f;

    stage_async(0);
    asm volatile("s_wait_asynccnt 0x0" ::: "memory");
    __syncthreads();

    unsigned curOff = 0;
    for (int k0 = 0; k0 < Ksz; k0 += 32) {
        const bool more = (k0 + 32) < Ksz;
        if (more) stage_async(curOff ^ BUFB);   // DMA next tile while we compute

        // ---- A fragments: lanes 0-15: K{0..7},{16..23}; 16-31: {8..15},{24..31}
        Frag fa[2];
#pragma unroll
        for (int i = 0; i < 2; ++i) {
            const bf16* ap = (const bf16*)((const char*)&As[0][32 * wm + 16 * i + l16][8 * half] + curOff);
            fa[i].q[0] = *(const B128*)ap;
            fa[i].q[1] = *(const B128*)(ap + 16);
        }
        // ---- B fragment per j, immediately consumed by both M strips
#pragma unroll
        for (int j = 0; j < 4; ++j) {
            Frag fb;
            const bf16* bp = (const bf16*)((const char*)&Bs[0][64 * wn + 16 * j + l16][8 * half] + curOff);
            fb.q[0] = *(const B128*)bp;
            fb.q[1] = *(const B128*)(bp + 16);
            acc[0][j] = __builtin_amdgcn_wmma_f32_16x16x32_bf16(
                false, fa[0].v, false, fb.v, (short)0, acc[0][j], false, false);
            acc[1][j] = __builtin_amdgcn_wmma_f32_16x16x32_bf16(
                false, fa[1].v, false, fb.v, (short)0, acc[1][j], false, false);
        }

        if (more) asm volatile("s_wait_asynccnt 0x0" ::: "memory");
        __syncthreads();
        curOff ^= BUFB;
    }

    // ---- store: VGPR r holds (M = r + 8*half, N = l16) per 16x16 C layout
#pragma unroll
    for (int i = 0; i < 2; ++i) {
        const int mBase = blockM + 32 * wm + 16 * i + half * 8;
#pragma unroll
        for (int j = 0; j < 4; ++j) {
            const int col = blockN + 64 * wn + 16 * j + l16;
#pragma unroll
            for (int r = 0; r < 8; ++r)
                C[(size_t)(mBase + r) * Nsz + col] = acc[i][j][r];
        }
    }
}

// ---------------- 2) chunked causal conv(k=4) + SiLU + local decay scan ----------------
// One thread per (b, chunk, d); 65536 threads. Chunk-local scan starts at h=0;
// chunk carries are combined in carry_kernel and applied in elemwise_y_kernel.
__global__ void conv_scan_kernel(const float* __restrict__ xz,      // [NROWS][2*DINNER]
                                 const float* __restrict__ conv_w,  // [DINNER][1][4]
                                 const float* __restrict__ conv_b,  // [DINNER]
                                 bf16* __restrict__ xconv_bf,       // [NROWS][DINNER]
                                 float* __restrict__ h_out,         // [NROWS][DINNER] (local)
                                 float* __restrict__ cend) {        // [BD][NCHUNK][DINNER]
    int idx = blockIdx.x * blockDim.x + threadIdx.x;
    if (idx >= BD * NCHUNK * DINNER) return;
    const int d     = idx % DINNER;
    const int rest  = idx / DINNER;
    const int chunk = rest % NCHUNK;
    const int b     = rest / NCHUNK;
    const int t0    = chunk * SCHUNK;

    const float w0 = conv_w[d * 4 + 0], w1 = conv_w[d * 4 + 1];
    const float w2 = conv_w[d * 4 + 2], w3 = conv_w[d * 4 + 3];
    const float bias = conv_b[d];

    const float* xin = xz + ((size_t)b * SD + t0) * (2 * DINNER) + d;
    float x0, x1, x2;
    if (chunk == 0) {
        x0 = 0.f; x1 = 0.f; x2 = 0.f;             // left zero-pad
    } else {
        const ptrdiff_t rs = (ptrdiff_t)(2 * DINNER);
        x0 = *(xin - 3 * rs);
        x1 = *(xin - 2 * rs);
        x2 = *(xin - 1 * rs);
    }

    float h = 0.f;
    for (int t = 0; t < SCHUNK; ++t) {
        float x3 = xin[(size_t)t * (2 * DINNER)];
        float c  = w0 * x0 + w1 * x1 + w2 * x2 + w3 * x3 + bias;
        float sc = silu_f(c);
        size_t o = ((size_t)b * SD + t0 + t) * DINNER + d;
        xconv_bf[o] = (bf16)sc;
        h = DECAYF * h + sc;
        h_out[o] = h;
        x0 = x1; x1 = x2; x2 = x3;
    }
    cend[((size_t)b * NCHUNK + chunk) * DINNER + d] = h;
}

// ---------------- 2b) combine chunk carries (tiny, sequential over NCHUNK) ----
__global__ void carry_kernel(const float* __restrict__ cend,
                             float* __restrict__ carry_in) {  // [BD][NCHUNK][DINNER]
    int idx = blockIdx.x * blockDim.x + threadIdx.x;
    if (idx >= BD * DINNER) return;
    const int d = idx % DINNER;
    const int b = idx / DINNER;
    const float kchunk = __powf(DECAYF, (float)SCHUNK);  // ~1.98e-6
    float carry = 0.f;
    for (int c = 0; c < NCHUNK; ++c) {
        size_t o = ((size_t)b * NCHUNK + c) * DINNER + d;
        carry_in[o] = carry;
        carry = kchunk * carry + cend[o];
    }
}

// ---------------- 3) s_C = x_conv @ W_x[16:32]^T   (wave per token row) ----------------
__launch_bounds__(256)
__global__ void sc_kernel(const bf16* __restrict__ xconv_bf, // [NROWS][DINNER]
                          const float* __restrict__ W_x,     // [2*DSTATE][DINNER]
                          float* __restrict__ sC) {          // [NROWS][DSTATE]
    const int wave = blockIdx.x * (blockDim.x >> 5) + (threadIdx.x >> 5);
    const int lane = threadIdx.x & 31;
    if (wave >= NROWS) return;

    const bf16* xr = xconv_bf + (size_t)wave * DINNER;
    float part[DSTATE];
#pragma unroll
    for (int n = 0; n < DSTATE; ++n) part[n] = 0.f;

    for (int k = lane; k < DINNER; k += 32) {
        float xv = (float)xr[k];
#pragma unroll
        for (int n = 0; n < DSTATE; ++n)
            part[n] += xv * W_x[(size_t)(DSTATE + n) * DINNER + k];
    }
#pragma unroll
    for (int n = 0; n < DSTATE; ++n) {
        float v = part[n];
        for (int off = 16; off > 0; off >>= 1)
            v += __shfl_down(v, off, 32);
        if (lane == 0) sC[(size_t)wave * DSTATE + n] = v;
    }
}

// ---------------- 4) y = (h_local + fixup) * (s_C . Bexp[d]) * silu(z) -> bf16 ----------
__launch_bounds__(256)
__global__ void elemwise_y_kernel(const float* __restrict__ h,        // chunk-local
                                  const float* __restrict__ carry_in, // [BD][NCHUNK][DINNER]
                                  const float* __restrict__ sC,
                                  const float* __restrict__ Bexp,     // [DINNER][DSTATE]
                                  const float* __restrict__ xz,       // [NROWS][2*DINNER]
                                  bf16* __restrict__ ybf) {           // [NROWS][DINNER]
    __shared__ float sCs[DSTATE];
    const int tilesPerRow = DINNER / 256;               // 8
    const int m     = blockIdx.x / tilesPerRow;
    const int dbase = (blockIdx.x % tilesPerRow) * 256;
    const int d     = dbase + threadIdx.x;
    const int b     = m / SD;
    const int t     = m % SD;
    const int chunk = t / SCHUNK;
    const int tl    = t % SCHUNK;

    if (threadIdx.x < DSTATE)
        sCs[threadIdx.x] = sC[(size_t)m * DSTATE + threadIdx.x];
    __syncthreads();

    const float4* bp = (const float4*)(Bexp + (size_t)d * DSTATE);
    float g = 0.f;
#pragma unroll
    for (int q = 0; q < 4; ++q) {
        float4 bv = bp[q];
        g += sCs[4 * q + 0] * bv.x + sCs[4 * q + 1] * bv.y +
             sCs[4 * q + 2] * bv.z + sCs[4 * q + 3] * bv.w;
    }
    size_t o  = (size_t)m * DINNER + d;
    float cin = carry_in[((size_t)b * NCHUNK + chunk) * DINNER + d];
    float hv  = h[o] + exp2f((float)(tl + 1) * LOG2DECAY) * cin;
    float zv  = xz[(size_t)m * (2 * DINNER) + DINNER + d];
    ybf[o] = (bf16)(hv * g * silu_f(zv));
}

// ---------------- 5) residual + LayerNorm, in place on d_out ----------------
__launch_bounds__(256)
__global__ void ln_kernel(const float* __restrict__ x,
                          const float* __restrict__ ln_g,
                          const float* __restrict__ ln_b,
                          float* __restrict__ out) {   // holds y2 on entry
    __shared__ float redS[8], redQ[8];
    const int m = blockIdx.x;
    const int t = threadIdx.x;

    float v[4];
    float sum = 0.f, sq = 0.f;
#pragma unroll
    for (int i = 0; i < 4; ++i) {
        int c = t + i * 256;
        float r = out[(size_t)m * DMODEL + c] + x[(size_t)m * DMODEL + c];
        v[i] = r; sum += r; sq += r * r;
    }
    for (int off = 16; off > 0; off >>= 1) {
        sum += __shfl_down(sum, off, 32);
        sq  += __shfl_down(sq,  off, 32);
    }
    const int wv = t >> 5, lane = t & 31;
    if (lane == 0) { redS[wv] = sum; redQ[wv] = sq; }
    __syncthreads();
    if (t == 0) {
        float s = 0.f, s2 = 0.f;
        for (int w = 0; w < 8; ++w) { s += redS[w]; s2 += redQ[w]; }
        redS[0] = s; redQ[0] = s2;
    }
    __syncthreads();
    const float mu   = redS[0] * (1.0f / DMODEL);
    const float var  = redQ[0] * (1.0f / DMODEL) - mu * mu;
    const float rstd = rsqrtf(var + LNEPS);
#pragma unroll
    for (int i = 0; i < 4; ++i) {
        int c = t + i * 256;
        out[(size_t)m * DMODEL + c] = (v[i] - mu) * rstd * ln_g[c] + ln_b[c];
    }
}

// ---------------- launch ----------------
extern "C" void kernel_launch(void* const* d_in, const int* in_sizes, int n_in,
                              void* d_out, int out_size, void* d_ws, size_t ws_size,
                              hipStream_t stream) {
    const float* x      = (const float*)d_in[0];
    const float* W_in   = (const float*)d_in[1];
    const float* conv_w = (const float*)d_in[2];
    const float* conv_b = (const float*)d_in[3];
    const float* W_x    = (const float*)d_in[4];
    const float* B_log  = (const float*)d_in[5];
    const float* W_out  = (const float*)d_in[6];
    const float* ln_g   = (const float*)d_in[7];
    const float* ln_b   = (const float*)d_in[8];
    float* out = (float*)d_out;

    // workspace carve (256B aligned)
    char* p = (char*)d_ws;
    auto carve = [&](size_t bytes) -> char* {
        char* r = p;
        p += (bytes + 255) & ~(size_t)255;
        return r;
    };
    bf16*  xbf   = (bf16*) carve((size_t)NROWS * DMODEL * 2);        // 16 MB
    bf16*  wibf  = (bf16*) carve((size_t)2 * DINNER * DMODEL * 2);   //  8 MB
    bf16*  wobf  = (bf16*) carve((size_t)DMODEL * DINNER * 2);       //  4 MB
    float* xz    = (float*)carve((size_t)NROWS * 2 * DINNER * 4);    // 128 MB
    bf16*  xcbf  = (bf16*) carve((size_t)NROWS * DINNER * 2);        // 32 MB
    float* hbuf  = (float*)carve((size_t)NROWS * DINNER * 4);        // 64 MB
    float* sC    = (float*)carve((size_t)NROWS * DSTATE * 4);
    float* Bexp  = (float*)carve((size_t)DINNER * DSTATE * 4);
    bf16*  ybf   = (bf16*) carve((size_t)NROWS * DINNER * 2);        // 32 MB
    float* cend  = (float*)carve((size_t)BD * NCHUNK * DINNER * 4);  // 256 KB
    float* cin   = (float*)carve((size_t)BD * NCHUNK * DINNER * 4);  // 256 KB

    // 0) precision staging
    {
        size_t n = (size_t)NROWS * DMODEL;
        cvt_f32_bf16_kernel<<<(unsigned)((n + 255) / 256), 256, 0, stream>>>(x, xbf, n);
    }
    {
        size_t n = (size_t)2 * DINNER * DMODEL;
        cvt_f32_bf16_kernel<<<(unsigned)((n + 255) / 256), 256, 0, stream>>>(W_in, wibf, n);
    }
    {
        size_t n = (size_t)DMODEL * DINNER;
        cvt_f32_bf16_kernel<<<(unsigned)((n + 255) / 256), 256, 0, stream>>>(W_out, wobf, n);
    }
    exp_kernel<<<(DINNER * DSTATE + 255) / 256, 256, 0, stream>>>(B_log, Bexp, DINNER * DSTATE);

    // 1) xz = x @ W_in^T   (M=8192, N=4096, K=1024)
    gemm_bf16_wmma_kernel<<<dim3((2 * DINNER) / 128, NROWS / 128), 256, 0, stream>>>(
        xbf, wibf, xz, NROWS, 2 * DINNER, DMODEL);

    // 2) chunked conv + silu + local scan, then carry combine
    conv_scan_kernel<<<(BD * NCHUNK * DINNER) / 256, 256, 0, stream>>>(
        xz, conv_w, conv_b, xcbf, hbuf, cend);
    carry_kernel<<<(BD * DINNER) / 256, 256, 0, stream>>>(cend, cin);

    // 3) s_C
    sc_kernel<<<NROWS / 8, 256, 0, stream>>>(xcbf, W_x, sC);

    // 4) elementwise y (applies scan carry fixup)
    elemwise_y_kernel<<<NROWS * (DINNER / 256), 256, 0, stream>>>(
        hbuf, cin, sC, Bexp, xz, ybf);

    // 5) y @ W_out^T -> d_out  (M=8192, N=1024, K=2048)
    gemm_bf16_wmma_kernel<<<dim3(DMODEL / 128, NROWS / 128), 256, 0, stream>>>(
        ybf, wobf, out, NROWS, DMODEL, DINNER);

    // 6) residual + LayerNorm (in place on d_out)
    ln_kernel<<<NROWS, 256, 0, stream>>>(x, ln_g, ln_b, out);
}
